// EncoderLayer_11132555231784
// MI455X (gfx1250) — matrix operand order repaired and checked by reading, not compiled, not run on hardware
//
#include <hip/hip_runtime.h>
#include <hip/hip_bf16.h>

#define Bdim 4
#define Ndim 2048
#define Kn   48
#define Cdim 128
#define INV_SCALE (1.0f / 30.0f)
#define EPSV 1e-5f

typedef __attribute__((ext_vector_type(16))) __bf16 v16bf;
typedef __attribute__((ext_vector_type(8)))  float  v8f;

union Frag { v16bf v; uint4 q[2]; unsigned short u[16]; };

__device__ __forceinline__ unsigned short f2bf(float f) {
  unsigned int u = __float_as_uint(f);
  u += 0x7FFFu + ((u >> 16) & 1u);          // round-to-nearest-even
  return (unsigned short)(u >> 16);
}

__device__ __forceinline__ float gelu_f(float x) {
  return 0.5f * x * (1.0f + erff(x * 0.70710678118654752f));
}

__device__ __forceinline__ v8f wmma_bf16(v16bf a, v16bf b, v8f c) {
  // D = A(16x32 bf16) * B(32x16 bf16) + C(16x16 f32)
  return __builtin_amdgcn_wmma_f32_16x16x32_bf16(false, a, false, b, (short)0, c,
                                                 false, false);
}

// Load one 16x32 bf16 A fragment from LDS row-major [rows][stride].
// ISA 16-bit A layout: lane<16 -> row M=lane,    K = {0..7, 16..23}
//                      lane>=16-> row M=lane-16, K = {8..15, 24..31}
__device__ __forceinline__ v16bf load_a(const unsigned short* base, int stride,
                                        int mt, int kt, int lane) {
  int row = mt * 16 + (lane & 15);
  int k0  = kt * 32 + ((lane >> 4) * 8);
  Frag f;
  f.q[0] = *(const uint4*)(base + row * stride + k0);
  f.q[1] = *(const uint4*)(base + row * stride + k0 + 16);
  return f.v;
}

// One MLP layer for this wave's 16-column tile: 3 M-tiles (48 rows).
// Packed weights: frag (kt, nt=w) is 32 lanes x 16 bf16 contiguous.
__device__ __forceinline__ void mlp_layer(const unsigned short* Asrc, int astride,
                                          int KT, const unsigned short* Wp,
                                          int w, int lane,
                                          v8f& a0, v8f& a1, v8f& a2) {
  a0 = {}; a1 = {}; a2 = {};
  for (int kt = 0; kt < KT; ++kt) {
    const v16bf bf = *(const v16bf*)(Wp + (size_t)(kt * 8 + w) * 512 + lane * 16);
    a0 = wmma_bf16(load_a(Asrc, astride, 0, kt, lane), bf, a0);
    a1 = wmma_bf16(load_a(Asrc, astride, 1, kt, lane), bf, a1);
    a2 = wmma_bf16(load_a(Asrc, astride, 2, kt, lane), bf, a2);
  }
}

// Store D tile (f32, +bias, gelu) as bf16 activations row-major [48][128].
// D layout: VGPR r -> row = mt*16 + (lane>=16 ? 8 : 0) + r, col = lane&15.
__device__ __forceinline__ void store_act(unsigned short* dst, int w, int lane,
                                          int mt, v8f v, float bias) {
  int col = w * 16 + (lane & 15);
  int rb  = mt * 16 + ((lane >> 4) * 8);
#pragma unroll
  for (int r = 0; r < 8; ++r) {
    dst[(rb + r) * 128 + col] = f2bf(gelu_f(v[r] + bias));
  }
}

// ---------------------------------------------------------------------------
// Pack f32 weight [Kdim][128] into WMMA B-fragment order as bf16.
// B layout: lanes 0-15 -> K {0..7, 16..23} (of 32-block), lanes 16-31 -> K {8..15, 24..31};
// col = nt*16 + (lane&15).
__global__ __launch_bounds__(256) void pack_weight(const float* __restrict__ W,
                                                   unsigned short* __restrict__ out,
                                                   int Kdim) {
  int tid = blockIdx.x * 256 + threadIdx.x;
  int total = Kdim * 128;
  if (tid >= total) return;
  int frag = tid >> 9;       // 512 elements per fragment
  int within = tid & 511;
  int lane = within >> 4, j = within & 15;
  int kt = frag >> 3, nt = frag & 7;
  int Kk = kt * 32 + ((lane >> 4) * 8) + (j & 7) + ((j >> 3) * 16);
  int Nn = nt * 16 + (lane & 15);
  out[tid] = f2bf(W[(size_t)Kk * 128 + Nn]);
}

__global__ __launch_bounds__(256) void f32_to_bf16_k(const float* __restrict__ in,
                                                     unsigned short* __restrict__ out,
                                                     int n) {
  int t = blockIdx.x * 256 + threadIdx.x;
  if (t < n) out[t] = f2bf(in[t]);
}

// ---------------------------------------------------------------------------
// Node update: message MLP over K neighbors, masked K-sum, LN, FFN, LN, mask.
__global__ __launch_bounds__(256) void node_kernel(
    const float* __restrict__ hV, const float* __restrict__ hE,
    const int* __restrict__ Eidx, const float* __restrict__ maskV,
    const float* __restrict__ maskAtt,
    const unsigned short* __restrict__ W1p, const float* __restrict__ b1,
    const unsigned short* __restrict__ W2p, const float* __restrict__ b2,
    const unsigned short* __restrict__ W3p, const float* __restrict__ b3,
    const float* __restrict__ Wffin, const float* __restrict__ bffin,
    const float* __restrict__ Wffout, const float* __restrict__ bffout,
    const float* __restrict__ g1, const float* __restrict__ be1,
    const float* __restrict__ g2, const float* __restrict__ be2,
    const unsigned short* __restrict__ hVb,
    float* __restrict__ hVout, unsigned short* __restrict__ hV2b) {
  __shared__ __align__(16) unsigned char smem[36864 + 12288 + 12288 + 64];
  unsigned short* A  = (unsigned short*)smem;             // [48][384] bf16
  unsigned short* H1 = (unsigned short*)(smem + 36864);   // [48][128] bf16
  unsigned short* H2 = (unsigned short*)(smem + 49152);   // [48][128] bf16
  // Aliases into A region (A is dead after layer 1):
  float* dh     = (float*)smem;              // [128]
  float* xv     = (float*)(smem + 512);      // [128]
  float* hv1    = (float*)(smem + 1024);     // [128]
  float* hidden = (float*)(smem + 2048);     // [512]
  float* red    = (float*)(smem + 4096);     // [2]: mean, rstd

  const int bn = blockIdx.x;                 // b*N + n
  const int tid = threadIdx.x;
  const int lane = tid & 31, w = tid >> 5;
  const int bb = bn / Ndim;

  // Build A = [hV_self | hE | hV[idx]] as bf16
  const unsigned short* selfrow = hVb + (size_t)bn * Cdim;
  const int* idxrow = Eidx + (size_t)bn * Kn;
  for (int e = tid; e < Kn * Cdim; e += 256) {
    int row = e >> 7, c = e & 127;
    A[row * 384 + c]       = selfrow[c];
    A[row * 384 + 128 + c] = f2bf(hE[((size_t)bn * Kn + row) * Cdim + c]);
    int nb = idxrow[row];
    A[row * 384 + 256 + c] = hVb[((size_t)bb * Ndim + nb) * Cdim + c];
  }
  __syncthreads();

  v8f a0, a1, a2;
  // Layer 1: [48,384] @ [384,128], gelu
  mlp_layer(A, 384, 12, W1p, w, lane, a0, a1, a2);
  float bias = b1[w * 16 + (lane & 15)];
  store_act(H1, w, lane, 0, a0, bias);
  store_act(H1, w, lane, 1, a1, bias);
  store_act(H1, w, lane, 2, a2, bias);
  __syncthreads();
  // Layer 2: [48,128] @ [128,128], gelu
  mlp_layer(H1, 128, 4, W2p, w, lane, a0, a1, a2);
  bias = b2[w * 16 + (lane & 15)];
  store_act(H2, w, lane, 0, a0, bias);
  store_act(H2, w, lane, 1, a1, bias);
  store_act(H2, w, lane, 2, a2, bias);
  __syncthreads();
  if (tid < 128) dh[tid] = 0.f;              // A region reused
  __syncthreads();
  // Layer 3: [48,128] @ [128,128]; bias, mask_attend, K-sum
  mlp_layer(H2, 128, 4, W3p, w, lane, a0, a1, a2);
  bias = b3[w * 16 + (lane & 15)];
  float colsum = 0.f;
  {
    const float* ma = maskAtt + (size_t)bn * Kn;
    int rbase = (lane >> 4) * 8;
#pragma unroll
    for (int r = 0; r < 8; ++r) colsum += (a0[r] + bias) * ma[rbase + r];
#pragma unroll
    for (int r = 0; r < 8; ++r) colsum += (a1[r] + bias) * ma[16 + rbase + r];
#pragma unroll
    for (int r = 0; r < 8; ++r) colsum += (a2[r] + bias) * ma[32 + rbase + r];
  }
  atomicAdd(&dh[w * 16 + (lane & 15)], colsum);
  __syncthreads();

  // LN1: x = hV + dh/SCALE
  if (tid < 128) xv[tid] = hV[(size_t)bn * Cdim + tid] + dh[tid] * INV_SCALE;
  __syncthreads();
  if (w == 0) {
    float s = 0.f, s2 = 0.f;
#pragma unroll
    for (int i = 0; i < 4; ++i) { float x = xv[lane + 32 * i]; s += x; s2 += x * x; }
    for (int m = 16; m >= 1; m >>= 1) { s += __shfl_xor(s, m, 32); s2 += __shfl_xor(s2, m, 32); }
    float mean = s * (1.f / 128.f);
    float var  = s2 * (1.f / 128.f) - mean * mean;
    if (lane == 0) { red[0] = mean; red[1] = rsqrtf(var + EPSV); }
  }
  __syncthreads();
  if (tid < 128) hv1[tid] = g1[tid] * (xv[tid] - red[0]) * red[1] + be1[tid];
  __syncthreads();

  // FFN: C -> 4C (gelu) -> C   (1.7% of FLOPs; f32 VALU)
  for (int j = tid; j < 512; j += 256) {
    float s = bffin[j];
    for (int c = 0; c < 128; ++c) s += hv1[c] * Wffin[(size_t)c * 512 + j];
    hidden[j] = gelu_f(s);
  }
  __syncthreads();
  if (tid < 128) {
    float s = bffout[tid];
    for (int j = 0; j < 512; ++j) s += hidden[j] * Wffout[(size_t)j * 128 + tid];
    xv[tid] = hv1[tid] + s;
  }
  __syncthreads();
  // LN2 + mask_V
  if (w == 0) {
    float s = 0.f, s2 = 0.f;
#pragma unroll
    for (int i = 0; i < 4; ++i) { float x = xv[lane + 32 * i]; s += x; s2 += x * x; }
    for (int m = 16; m >= 1; m >>= 1) { s += __shfl_xor(s, m, 32); s2 += __shfl_xor(s2, m, 32); }
    float mean = s * (1.f / 128.f);
    float var  = s2 * (1.f / 128.f) - mean * mean;
    if (lane == 0) { red[0] = mean; red[1] = rsqrtf(var + EPSV); }
  }
  __syncthreads();
  if (tid < 128) {
    float y = (g2[tid] * (xv[tid] - red[0]) * red[1] + be2[tid]) * maskV[bn];
    hVout[(size_t)bn * Cdim + tid] = y;
    hV2b[(size_t)bn * Cdim + tid] = f2bf(y);
  }
}

// ---------------------------------------------------------------------------
// Edge update: same MLP with updated h_V, then per-edge LayerNorm.
__global__ __launch_bounds__(256) void edge_kernel(
    const float* __restrict__ hE, const int* __restrict__ Eidx,
    const unsigned short* __restrict__ W11p, const float* __restrict__ b11,
    const unsigned short* __restrict__ W12p, const float* __restrict__ b12,
    const unsigned short* __restrict__ W13p, const float* __restrict__ b13,
    const float* __restrict__ g3, const float* __restrict__ be3,
    const unsigned short* __restrict__ hV2b, float* __restrict__ hEout) {
  __shared__ __align__(16) unsigned char smem[36864 + 12288 + 12288 + 64];
  unsigned short* A  = (unsigned short*)smem;
  unsigned short* H1 = (unsigned short*)(smem + 36864);
  unsigned short* H2 = (unsigned short*)(smem + 49152);
  float* res = (float*)smem;                  // [48][128] f32 (A region reuse)

  const int bn = blockIdx.x;
  const int tid = threadIdx.x;
  const int lane = tid & 31, w = tid >> 5;
  const int bb = bn / Ndim;

  const unsigned short* selfrow = hV2b + (size_t)bn * Cdim;
  const int* idxrow = Eidx + (size_t)bn * Kn;
  for (int e = tid; e < Kn * Cdim; e += 256) {
    int row = e >> 7, c = e & 127;
    A[row * 384 + c]       = selfrow[c];
    A[row * 384 + 128 + c] = f2bf(hE[((size_t)bn * Kn + row) * Cdim + c]);
    int nb = idxrow[row];
    A[row * 384 + 256 + c] = hV2b[((size_t)bb * Ndim + nb) * Cdim + c];
  }
  __syncthreads();

  v8f a0, a1, a2;
  mlp_layer(A, 384, 12, W11p, w, lane, a0, a1, a2);
  float bias = b11[w * 16 + (lane & 15)];
  store_act(H1, w, lane, 0, a0, bias);
  store_act(H1, w, lane, 1, a1, bias);
  store_act(H1, w, lane, 2, a2, bias);
  __syncthreads();
  mlp_layer(H1, 128, 4, W12p, w, lane, a0, a1, a2);
  bias = b12[w * 16 + (lane & 15)];
  store_act(H2, w, lane, 0, a0, bias);
  store_act(H2, w, lane, 1, a1, bias);
  store_act(H2, w, lane, 2, a2, bias);
  __syncthreads();
  mlp_layer(H2, 128, 4, W13p, w, lane, a0, a1, a2);
  bias = b13[w * 16 + (lane & 15)];
  {
    int col = w * 16 + (lane & 15);
    int rbase = (lane >> 4) * 8;
#pragma unroll
    for (int r = 0; r < 8; ++r) {
      int row = rbase + r;
      res[row * 128 + col] =
          hE[((size_t)bn * Kn + row) * Cdim + col] + a0[r] + bias;
    }
#pragma unroll
    for (int r = 0; r < 8; ++r) {
      int row = 16 + rbase + r;
      res[row * 128 + col] =
          hE[((size_t)bn * Kn + row) * Cdim + col] + a1[r] + bias;
    }
#pragma unroll
    for (int r = 0; r < 8; ++r) {
      int row = 32 + rbase + r;
      res[row * 128 + col] =
          hE[((size_t)bn * Kn + row) * Cdim + col] + a2[r] + bias;
    }
  }
  __syncthreads();

  // Per-edge LayerNorm over C: each wave handles 6 rows
  for (int rr = 0; rr < 6; ++rr) {
    int row = w * 6 + rr;
    float x0 = res[row * 128 + lane];
    float x1 = res[row * 128 + lane + 32];
    float x2 = res[row * 128 + lane + 64];
    float x3 = res[row * 128 + lane + 96];
    float s = x0 + x1 + x2 + x3;
    float s2 = x0 * x0 + x1 * x1 + x2 * x2 + x3 * x3;
    for (int m = 16; m >= 1; m >>= 1) { s += __shfl_xor(s, m, 32); s2 += __shfl_xor(s2, m, 32); }
    float mean = s * (1.f / 128.f);
    float rstd = rsqrtf(s2 * (1.f / 128.f) - mean * mean + EPSV);
    size_t base = ((size_t)bn * Kn + row) * Cdim;
    hEout[base + lane]      = g3[lane]      * (x0 - mean) * rstd + be3[lane];
    hEout[base + lane + 32] = g3[lane + 32] * (x1 - mean) * rstd + be3[lane + 32];
    hEout[base + lane + 64] = g3[lane + 64] * (x2 - mean) * rstd + be3[lane + 64];
    hEout[base + lane + 96] = g3[lane + 96] * (x3 - mean) * rstd + be3[lane + 96];
  }
}

// ---------------------------------------------------------------------------
extern "C" void kernel_launch(void* const* d_in, const int* in_sizes, int n_in,
                              void* d_out, int out_size, void* d_ws, size_t ws_size,
                              hipStream_t stream) {
  const float* hV      = (const float*)d_in[0];
  const float* hE      = (const float*)d_in[1];
  const int*   Eidx    = (const int*)d_in[2];
  const float* maskV   = (const float*)d_in[3];
  const float* maskAtt = (const float*)d_in[4];
  const float* W1  = (const float*)d_in[5];  const float* b1  = (const float*)d_in[6];
  const float* W2  = (const float*)d_in[7];  const float* b2  = (const float*)d_in[8];
  const float* W3  = (const float*)d_in[9];  const float* b3  = (const float*)d_in[10];
  const float* W11 = (const float*)d_in[11]; const float* b11 = (const float*)d_in[12];
  const float* W12 = (const float*)d_in[13]; const float* b12 = (const float*)d_in[14];
  const float* W13 = (const float*)d_in[15]; const float* b13 = (const float*)d_in[16];
  const float* Wffin  = (const float*)d_in[17]; const float* bffin  = (const float*)d_in[18];
  const float* Wffout = (const float*)d_in[19]; const float* bffout = (const float*)d_in[20];
  const float* g1 = (const float*)d_in[21]; const float* be1 = (const float*)d_in[22];
  const float* g2 = (const float*)d_in[23]; const float* be2 = (const float*)d_in[24];
  const float* g3 = (const float*)d_in[25]; const float* be3 = (const float*)d_in[26];

  unsigned char* ws = (unsigned char*)d_ws;
  unsigned short* W1p  = (unsigned short*)(ws + 0);        //  98304 B
  unsigned short* W2p  = (unsigned short*)(ws + 98304);    //  32768 B
  unsigned short* W3p  = (unsigned short*)(ws + 131072);   //  32768 B
  unsigned short* W11p = (unsigned short*)(ws + 163840);   //  98304 B
  unsigned short* W12p = (unsigned short*)(ws + 262144);   //  32768 B
  unsigned short* W13p = (unsigned short*)(ws + 294912);   //  32768 B
  unsigned short* hVb  = (unsigned short*)(ws + 327680);   // 2 MiB
  unsigned short* hV2b = (unsigned short*)(ws + 2424832);  // 2 MiB

  float* hVout = (float*)d_out;
  float* hEout = (float*)d_out + (size_t)Bdim * Ndim * Cdim;

  pack_weight<<<(384 * 128 + 255) / 256, 256, 0, stream>>>(W1, W1p, 384);
  pack_weight<<<(128 * 128 + 255) / 256, 256, 0, stream>>>(W2, W2p, 128);
  pack_weight<<<(128 * 128 + 255) / 256, 256, 0, stream>>>(W3, W3p, 128);
  pack_weight<<<(384 * 128 + 255) / 256, 256, 0, stream>>>(W11, W11p, 384);
  pack_weight<<<(128 * 128 + 255) / 256, 256, 0, stream>>>(W12, W12p, 128);
  pack_weight<<<(128 * 128 + 255) / 256, 256, 0, stream>>>(W13, W13p, 128);
  int nhv = Bdim * Ndim * Cdim;
  f32_to_bf16_k<<<(nhv + 255) / 256, 256, 0, stream>>>(hV, hVb, nhv);

  node_kernel<<<Bdim * Ndim, 256, 0, stream>>>(
      hV, hE, Eidx, maskV, maskAtt, W1p, b1, W2p, b2, W3p, b3,
      Wffin, bffin, Wffout, bffout, g1, be1, g2, be2, hVb, hVout, hV2b);

  edge_kernel<<<Bdim * Ndim, 256, 0, stream>>>(
      hE, Eidx, W11p, b11, W12p, b12, W13p, b13, g3, be3, hV2b, hEout);
}
